// CapsNet_84817014161436
// MI455X (gfx1250) — compile-verified, matrix operand order checked by reading, not run on hardware
//
#include <hip/hip_runtime.h>
#include <math.h>

// CapsNet forward for MI455X (gfx1250, wave32).
// conv2 / primary-caps / u_hat run on v_wmma_f32_16x16x32_bf16 with f32 accum.
// Activations are bf16 NHWC; im2col K is reordered to (ky,kx,ic) so every WMMA
// operand is two contiguous 16B loads (no per-element division/gather).
// Primary-caps input image (102.4 KB) is staged into LDS with the TDM
// (tensor_load_to_lds + s_wait_tensorcnt) when the builtin is available.

typedef __bf16 bf16;
typedef __attribute__((ext_vector_type(16))) __bf16 v16bf;
typedef __attribute__((ext_vector_type(8)))  __bf16 v8bf;
typedef __attribute__((ext_vector_type(8)))  float  v8f;

#if defined(__HIP_DEVICE_COMPILE__) && __has_builtin(__builtin_amdgcn_tensor_load_to_lds)
#define USE_TDM 1
typedef unsigned int tdm_v4u __attribute__((ext_vector_type(4)));
typedef int          tdm_v4i __attribute__((ext_vector_type(4)));
typedef int          tdm_v8i __attribute__((ext_vector_type(8)));
#else
#define USE_TDM 0
#endif

// Concatenate two 16B bf16 chunks into one WMMA operand.
// 16-bit operand layout (ISA 7.12.2): element j holds K = klo + (j<8 ? j : 16+j-8),
// klo = 8*(lane>>4)  ->  two contiguous runs of 8 at +0 and +16 elements.
__device__ __forceinline__ v16bf ld_op(const bf16* p) {
  v8bf lo = *(const v8bf*)p;
  v8bf hi = *(const v8bf*)(p + 16);
  return __builtin_shufflevector(lo, hi, 0,1,2,3,4,5,6,7,8,9,10,11,12,13,14,15);
}

// ---------------------------------------------------------------- converts
__global__ void f32_to_bf16_kernel(const float* __restrict__ src,
                                   bf16* __restrict__ dst, int n) {
  int i = blockIdx.x * blockDim.x + threadIdx.x;
  if (i < n) dst[i] = (bf16)src[i];
}

// conv2_w [128,64,3,3] f32 -> [128][kk=ky*3+kx][ic] bf16
__global__ void reorder_w2_kernel(const float* __restrict__ src, bf16* __restrict__ dst) {
  int i = blockIdx.x * blockDim.x + threadIdx.x;     // 73728
  if (i >= 73728) return;
  const int ic = i & 63, kk = (i >> 6) % 9, oc = i / 576;
  const int ky = kk / 3, kx = kk % 3;
  dst[i] = (bf16)src[((oc * 64 + ic) * 3 + ky) * 3 + kx];
}

// prim_w [256,128,9,9] f32 -> [256][kk=ky*9+kx][ic] bf16
__global__ void reorder_pw_kernel(const float* __restrict__ src, bf16* __restrict__ dst) {
  int i = blockIdx.x * blockDim.x + threadIdx.x;     // 2654208
  if (i >= 2654208) return;
  const int ic = i & 127, kk = (i >> 7) % 81, oc = i / 10368;
  dst[i] = (bf16)src[(oc * 128 + ic) * 81 + kk];
}

__global__ void zero_kernel(float* __restrict__ p, int n) {
  int i = blockIdx.x * blockDim.x + threadIdx.x;
  if (i < n) p[i] = 0.0f;
}

// ------------------------------------------- conv1 + relu + maxpool(pad1)
// data [256,3,76,76] -> pool1 bf16 NHWC [256,38,38,64]. K=27: VALU, mem bound.
// 256 threads = 4 positions x 64 channels -> coalesced 128B NHWC stores.
__global__ void conv1_relu_pool_kernel(const float* __restrict__ x,
                                       const float* __restrict__ w,
                                       const float* __restrict__ bias,
                                       bf16* __restrict__ out) {
  __shared__ float wk[27 * 64];                      // [k][oc]
  __shared__ float bs[64];
  for (int i = threadIdx.x; i < 64 * 27; i += 256) {
    const int oc = i / 27, t = i % 27;
    wk[t * 64 + oc] = w[i];
  }
  if (threadIdx.x < 64) bs[threadIdx.x] = bias[threadIdx.x];
  __syncthreads();
  const int oc = threadIdx.x & 63;
  const int b  = blockIdx.y;
  const int p  = blockIdx.x * 4 + (threadIdx.x >> 6);   // 361*4 = 1444
  const int py = p / 38, px = p % 38;
  float m = -INFINITY;
#pragma unroll
  for (int dy = 0; dy < 2; ++dy) {
    const int cy = 2 * py - 1 + dy;                   // conv row (74), pool pad 1
    if (cy < 0 || cy >= 74) continue;
#pragma unroll
    for (int dx = 0; dx < 2; ++dx) {
      const int cx = 2 * px - 1 + dx;
      if (cx < 0 || cx >= 74) continue;
      float acc = bs[oc];
#pragma unroll
      for (int ic = 0; ic < 3; ++ic)
#pragma unroll
        for (int ky = 0; ky < 3; ++ky)
#pragma unroll
          for (int kx = 0; kx < 3; ++kx)
            acc = fmaf(x[((b * 3 + ic) * 76 + (cy + ky)) * 76 + (cx + kx)],
                       wk[(ic * 9 + ky * 3 + kx) * 64 + oc], acc);
      m = fmaxf(m, fmaxf(acc, 0.0f));
    }
  }
  out[(size_t)((b * 38 + py) * 38 + px) * 64 + oc] = (bf16)m;
}

// --------------------------------------------- conv2 implicit-GEMM (WMMA)
// pool1 NHWC [256,38,38,64] (pad 1) x w2 [128,9,64] -> c2out NHWC [256,1444,128]
// Workgroup = (batch, output row): LDS halo 3x40x64 bf16 (15 KB, zero padded).
// M = 38 positions (3 tiles), N = 128 ch (8 waves), K = 576 (9 kk x 2 halves).
__global__ void conv2_wmma_kernel(const bf16* __restrict__ xin,
                                  const bf16* __restrict__ wb,
                                  const float* __restrict__ bias,
                                  bf16* __restrict__ out) {
  __shared__ bf16 halo[3 * 40 * 64];
  const int oy = blockIdx.x;                         // 0..37
  const int b  = blockIdx.y;                         // 0..255
  {
    uint4* dv = (uint4*)halo;
    for (int i = threadIdx.x; i < 3 * 40 * 8; i += 256) {
      const int part = i & 7;
      const int cell = i >> 3;                       // 0..119
      const int dy = cell / 40;
      const int xx = (cell % 40) - 1;                // halo col -1..38
      const int iy = oy - 1 + dy;
      uint4 v = make_uint4(0, 0, 0, 0);
      if (iy >= 0 && iy < 38 && xx >= 0 && xx < 38)
        v = ((const uint4*)xin)[(size_t)((b * 38 + iy) * 38 + xx) * 8 + part];
      dv[i] = v;
    }
  }
  __syncthreads();
  const int wave = threadIdx.x >> 5;                 // oc tile
  const int lane = threadIdx.x & 31;
  const int col  = lane & 15;
  const int klo  = (lane >> 4) << 3;                 // 0 or 8
  const int oc   = wave * 16 + col;
  v8f acc[3] = {};
  const bf16* wbase = wb + (size_t)oc * 576 + klo;
  for (int kk = 0; kk < 9; ++kk) {
    const int ky = kk / 3, kx = kk % 3;              // tiny, hoisted by unroll below
    const int hoff = (ky * 40 + kx) * 64 + klo;      // wave-uniform per chunk
#pragma unroll
    for (int h = 0; h < 2; ++h) {
      v16bf bv = ld_op(wbase + kk * 64 + h * 32);
#pragma unroll
      for (int mt = 0; mt < 3; ++mt) {
        const int ox  = mt * 16 + col;
        const int oxe = (ox < 38) ? ox : 0;          // clamp; rows unwritten anyway
        v16bf av = ld_op(halo + hoff + oxe * 64 + h * 32);
        acc[mt] = __builtin_amdgcn_wmma_f32_16x16x32_bf16(
            false, av, false, bv, (short)0, acc[mt], false, false);
      }
    }
  }
  const float bvl = bias[oc];
#pragma unroll
  for (int mt = 0; mt < 3; ++mt)
#pragma unroll
    for (int r = 0; r < 8; ++r) {
      const int ox = mt * 16 + ((lane >> 4) << 3) + r;
      if (ox < 38)
        out[((size_t)b * 1444 + oy * 38 + ox) * 128 + oc] =
            (bf16)fmaxf(acc[mt][r] + bvl, 0.0f);
    }
}

// ------------------------------------------------ maxpool2(pad1) 38 -> 20
// c2out NHWC [256,1444,128] -> pool2 NHWC [256,400,128]
__global__ void pool2_kernel(const bf16* __restrict__ cin, bf16* __restrict__ out) {
  const int idx = blockIdx.x * blockDim.x + threadIdx.x;   // 256*400*128
  if (idx >= 256 * 400 * 128) return;
  const int c = idx & 127;
  int t = idx >> 7;
  const int px = t % 20; t /= 20;
  const int py = t % 20; const int b = t / 20;
  float m = -INFINITY;
  for (int dy = 0; dy < 2; ++dy) {
    const int iy = 2 * py - 1 + dy; if (iy < 0 || iy >= 38) continue;
    for (int dx = 0; dx < 2; ++dx) {
      const int ix = 2 * px - 1 + dx; if (ix < 0 || ix >= 38) continue;
      m = fmaxf(m, (float)cin[((size_t)b * 1444 + iy * 38 + ix) * 128 + c]);
    }
  }
  out[idx] = (bf16)m;
}

// ------------------------------------- primary caps implicit-GEMM (WMMA)
// pool2 NHWC [256,400,128] (TDM-staged in LDS) x pw [256,81,128]
// -> upre f32 [256,9216].  M=36 pos, N=256 ch (8 waves x 2 tiles), K=10368.
__global__ void primcaps_wmma_kernel(const bf16* __restrict__ xin,
                                     const bf16* __restrict__ wb,
                                     const float* __restrict__ bias,
                                     float* __restrict__ upre) {
  extern __shared__ bf16 smem[];                     // 400*128 bf16 = 102400 B
  const int mt = blockIdx.x;                         // 0..2
  const int b  = blockIdx.y;                         // 0..255
#if USE_TDM
  if (threadIdx.x < 32) {                            // wave 0 issues the TDM op
    const unsigned long long ga = (unsigned long long)(const void*)(xin + (size_t)b * 51200);
    const unsigned lds = (unsigned)(size_t)smem;
    const unsigned n8 = 12800;                       // 102400 B in 8B units
    tdm_v4u g0 = {};
    g0[0] = 1u;                                      // count=1, user mode
    g0[1] = lds;                                     // lds_addr
    g0[2] = (unsigned)ga;                            // global_addr[31:0]
    g0[3] = ((unsigned)(ga >> 32) & 0x01FFFFFFu) | 0x80000000u;  // addr[56:32] | type=2
    tdm_v8i g1 = {};
    g1[0] = (int)(3u << 16);                         // data_size=3 (8B), no flags
    g1[1] = (int)((n8 & 0xFFFFu) << 16);             // tensor_dim0 lo16 @ [63:48]
    g1[2] = (int)((n8 >> 16) | (1u << 16));          // tensor_dim0 hi16 | tensor_dim1=1
    g1[3] = (int)(n8 << 16);                         // tile_dim0 @ [127:112]
    g1[4] = 0;                                       // tile_dim1=0 (1-D), tile_dim2=0
    g1[5] = (int)n8;                                 // tensor_dim0_stride lo32
    g1[6] = 0; g1[7] = 0;
    tdm_v4i gz4 = {};
#if defined(__clang_major__) && __clang_major__ >= 23
    tdm_v8i gz8 = {};
    __builtin_amdgcn_tensor_load_to_lds(g0, g1, gz4, gz4, gz8, 0);
#else
    __builtin_amdgcn_tensor_load_to_lds(g0, g1, gz4, gz4, 0);
#endif
    __builtin_amdgcn_s_wait_tensorcnt(0);
  }
#else
  {
    const uint4* src = (const uint4*)(xin + (size_t)b * 51200);
    uint4* dst = (uint4*)smem;
    for (int i = threadIdx.x; i < 6400; i += 256) dst[i] = src[i];
  }
#endif
  __syncthreads();
  const int wave = threadIdx.x >> 5;
  const int lane = threadIdx.x & 31;
  const int col  = lane & 15;
  const int klo  = (lane >> 4) << 3;
  const int p  = mt * 16 + col;
  const int pp = (p < 36) ? p : 0;                   // clamp; rows unwritten
  const int oy = pp / 6, ox = pp % 6;
  const int abase = (oy * 40 + ox * 2) * 128 + klo;  // NHWC, stride-2 window
  const int n0 = wave, n1 = wave + 8;
  const bf16* b0p = wb + (size_t)(n0 * 16 + col) * 10368 + klo;
  const bf16* b1p = wb + (size_t)(n1 * 16 + col) * 10368 + klo;
  v8f acc0 = {}, acc1 = {};
  for (int kk = 0; kk < 81; ++kk) {
    const int aoff = abase + ((kk / 9) * 20 + (kk % 9)) * 128;   // uniform shift
    __builtin_prefetch(b0p + (kk + 1) * 128, 0, 1);
#pragma unroll
    for (int h = 0; h < 4; ++h) {
      v16bf av  = ld_op(smem + aoff + h * 32);
      v16bf bv0 = ld_op(b0p + kk * 128 + h * 32);
      v16bf bv1 = ld_op(b1p + kk * 128 + h * 32);
      acc0 = __builtin_amdgcn_wmma_f32_16x16x32_bf16(
          false, av, false, bv0, (short)0, acc0, false, false);
      acc1 = __builtin_amdgcn_wmma_f32_16x16x32_bf16(
          false, av, false, bv1, (short)0, acc1, false, false);
    }
  }
#pragma unroll
  for (int t = 0; t < 2; ++t) {
    const int oc = (t ? n1 : n0) * 16 + col;
    const float bvl = bias[oc];
#pragma unroll
    for (int r = 0; r < 8; ++r) {
      const int pm = mt * 16 + ((lane >> 4) << 3) + r;
      const float v = (t ? acc1[r] : acc0[r]) + bvl;
      if (pm < 36) upre[(size_t)b * 9216 + oc * 36 + pm] = v;
    }
  }
}

// ------------------------------------------ squash over capsule dim (I=8)
__global__ void squash_u_kernel(float* __restrict__ u, bf16* __restrict__ ubf) {
  const int idx = blockIdx.x * blockDim.x + threadIdx.x;   // 256*1152
  if (idx >= 256 * 1152) return;
  float t[8], sn = 0.0f;
#pragma unroll
  for (int i = 0; i < 8; ++i) { t[i] = u[idx * 8 + i]; sn = fmaf(t[i], t[i], sn); }
  const float scale = sn / ((1.0f + sn) * sqrtf(sn));
#pragma unroll
  for (int i = 0; i < 8; ++i) {
    const float v = t[i] * scale;
    u[idx * 8 + i]   = v;
    ubf[idx * 8 + i] = (bf16)v;
  }
}

// -------------------------------- u_hat = einsum(rcoi,bri->brco) via WMMA
// Per route r: GEMM M=256 batches, N=160 (c*16+o), K=8 (upper K lanes zero).
__global__ void uhat_wmma_kernel(const bf16* __restrict__ ubf,   // [256,1152,8]
                                 const bf16* __restrict__ Wbf,   // [1152,160,8]
                                 float* __restrict__ uhat) {     // [256,1152,160]
  const int mt   = blockIdx.x;                       // batch tile 0..15
  const int rr   = blockIdx.y;                       // route 0..1151
  const int nt   = threadIdx.x >> 5;                 // 0..9 (320 threads)
  const int lane = threadIdx.x & 31;
  const int col  = lane & 15;
  const int bidx = mt * 16 + col;
  v16bf a = {}, bm = {};
  if (lane < 16) {                                   // K=0..7 live only here
    v8bf ua = *(const v8bf*)(ubf + (size_t)(bidx * 1152 + rr) * 8);
    v8bf wv = *(const v8bf*)(Wbf + (size_t)(rr * 160 + nt * 16 + col) * 8);
#pragma unroll
    for (int j = 0; j < 8; ++j) { a[j] = ua[j]; bm[j] = wv[j]; }
  }
  v8f acc = {};
  acc = __builtin_amdgcn_wmma_f32_16x16x32_bf16(false, a, false, bm,
                                                (short)0, acc, false, false);
  const int n = nt * 16 + col;
#pragma unroll
  for (int r = 0; r < 8; ++r) {
    const int bb = mt * 16 + ((lane >> 4) << 3) + r;
    uhat[((size_t)bb * 1152 + rr) * 160 + n] = acc[r];
  }
}

// ------------------------------------------ routing: softmax over routes
__global__ void softmax_routes_kernel(const float* __restrict__ bij,
                                      float* __restrict__ cij) {
  const int c = blockIdx.x;                          // 0..9
  __shared__ float red[256];
  float mx = -INFINITY;
  for (int r = threadIdx.x; r < 1152; r += 256) mx = fmaxf(mx, bij[r * 10 + c]);
  red[threadIdx.x] = mx; __syncthreads();
  for (int s = 128; s > 0; s >>= 1) {
    if (threadIdx.x < s) red[threadIdx.x] = fmaxf(red[threadIdx.x], red[threadIdx.x + s]);
    __syncthreads();
  }
  mx = red[0]; __syncthreads();
  float sum = 0.0f;
  for (int r = threadIdx.x; r < 1152; r += 256) sum += __expf(bij[r * 10 + c] - mx);
  red[threadIdx.x] = sum; __syncthreads();
  for (int s = 128; s > 0; s >>= 1) {
    if (threadIdx.x < s) red[threadIdx.x] += red[threadIdx.x + s];
    __syncthreads();
  }
  const float inv = 1.0f / red[0];
  for (int r = threadIdx.x; r < 1152; r += 256)
    cij[r * 10 + c] = __expf(bij[r * 10 + c] - mx) * inv;
}

// s_j over routes + elementwise squash: v = s*|s|/(1+s^2)
__global__ void route_sv_kernel(const float* __restrict__ uhat,
                                const float* __restrict__ cij,
                                float* __restrict__ vout) {      // [256,10,16]
  const int idx = blockIdx.x * blockDim.x + threadIdx.x;   // 256*160
  if (idx >= 256 * 160) return;
  const int b = idx / 160, n = idx % 160, c = n / 16;
  const float* up = uhat + (size_t)b * 1152 * 160 + n;
  float s = 0.0f;
  for (int r = 0; r < 1152; ++r) s = fmaf(cij[r * 10 + c], up[(size_t)r * 160], s);
  vout[idx] = s * fabsf(s) / (1.0f + s * s);
}

// b_ij += mean_b( <u_hat[b,r,c,:], v[b,c,:]> )
__global__ void route_bupdate_kernel(const float* __restrict__ uhat,
                                     const float* __restrict__ v,
                                     float* __restrict__ bij) {
  const int idx = blockIdx.x * blockDim.x + threadIdx.x;   // 1152*10
  if (idx >= 11520) return;
  const int r = idx / 10, c = idx % 10;
  float a = 0.0f;
  for (int b = 0; b < 256; ++b) {
    const float* up = uhat + ((size_t)(b * 1152 + r) * 10 + c) * 16;
    const float* vp = v + (b * 10 + c) * 16;
    float d = 0.0f;
#pragma unroll
    for (int o = 0; o < 16; ++o) d = fmaf(up[o], vp[o], d);
    a += d;
  }
  bij[idx] += a * (1.0f / 256.0f);
}

// --------------------- 10 small classifiers: 160->100 relu -> 1 sigmoid
__global__ void classifier_kernel(const float* __restrict__ feat,   // [256,160]
                                  const float* __restrict__ w1,     // [10,160,100]
                                  const float* __restrict__ b1,     // [10,100]
                                  const float* __restrict__ w2,     // [10,100]
                                  const float* __restrict__ b2,     // [10]
                                  float* __restrict__ cls) {        // [256,10]
  __shared__ float h[100];
  const int b = blockIdx.x / 10, k = blockIdx.x % 10;
  const float* f = feat + b * 160;
  for (int j = threadIdx.x; j < 100; j += blockDim.x) {
    float acc = b1[k * 100 + j];
    for (int i = 0; i < 160; ++i) acc = fmaf(f[i], w1[(k * 160 + i) * 100 + j], acc);
    h[j] = fmaxf(acc, 0.0f);
  }
  __syncthreads();
  if (threadIdx.x == 0) {
    float acc = b2[k];
    for (int j = 0; j < 100; ++j) acc = fmaf(h[j], w2[k * 100 + j], acc);
    cls[b * 10 + k] = 1.0f / (1.0f + __expf(-acc));
  }
}

// ---------------------------------------------------------------- driver
extern "C" void kernel_launch(void* const* d_in, const int* in_sizes, int n_in,
                              void* d_out, int out_size, void* d_ws, size_t ws_size,
                              hipStream_t stream) {
  const float* data    = (const float*)d_in[0];
  const float* conv1_w = (const float*)d_in[1];
  const float* conv1_b = (const float*)d_in[2];
  const float* conv2_w = (const float*)d_in[3];
  const float* conv2_b = (const float*)d_in[4];
  const float* prim_w  = (const float*)d_in[5];
  const float* prim_b  = (const float*)d_in[6];
  const float* Wf      = (const float*)d_in[7];
  const float* cls_w1  = (const float*)d_in[8];
  const float* cls_b1  = (const float*)d_in[9];
  const float* cls_w2  = (const float*)d_in[10];
  const float* cls_b2  = (const float*)d_in[11];

  // Workspace layout (bytes, 256-aligned). u_hat (188.7 MB) aliases
  // pool1 + conv2out (141.9 MB) — lifetimes disjoint. Peak ~237.6 MB.
  char* ws = (char*)d_ws;
  float* uhat  = (float*)(ws + 0);                 // [256,1152,160] f32
  bf16*  pool1 = (bf16*)(ws + 0);                  // NHWC [256,38,38,64]
  bf16*  c2out = (bf16*)(ws + 47316992ull);        // NHWC [256,1444,128]
  bf16*  pool2 = (bf16*)(ws + 188743680ull);       // NHWC [256,400,128]
  float* u     = (float*)(ws + 214958080ull);      // [256,1152,8] f32
  bf16*  ubf   = (bf16*)(ws + 224395264ull);       // [256,1152,8]
  bf16*  w2p   = (bf16*)(ws + 229113856ull);       // [128,9,64]
  bf16*  pwp   = (bf16*)(ws + 229261312ull);       // [256,81,128]
  bf16*  Wbf   = (bf16*)(ws + 234569728ull);       // [1152,160,8]
  float* bij   = (float*)(ws + 237518848ull);      // [1152,10]
  float* cij   = (float*)(ws + 237564928ull);      // [1152,10]

  float* vout = (float*)d_out;                     // output[0]: [256,10,16,1]
  float* cls  = vout + 40960;                      // output[1]: [256,10]

  // weight reorder/convert to bf16 (K order = ky,kx,ic)
  reorder_w2_kernel<<<dim3(288), 256, 0, stream>>>(conv2_w, w2p);
  reorder_pw_kernel<<<dim3(10368), 256, 0, stream>>>(prim_w, pwp);
  f32_to_bf16_kernel<<<dim3(5760), 256, 0, stream>>>(Wf, Wbf, 1474560);

  conv1_relu_pool_kernel<<<dim3(361, 256), 256, 0, stream>>>(data, conv1_w, conv1_b, pool1);
  conv2_wmma_kernel<<<dim3(38, 256), 256, 0, stream>>>(pool1, w2p, conv2_b, c2out);
  pool2_kernel<<<dim3(51200), 256, 0, stream>>>(c2out, pool2);
  primcaps_wmma_kernel<<<dim3(3, 256), 256, 102400, stream>>>(pool2, pwp, prim_b, u);
  squash_u_kernel<<<dim3(1152), 256, 0, stream>>>(u, ubf);
  uhat_wmma_kernel<<<dim3(16, 1152), 320, 0, stream>>>(ubf, Wbf, uhat);

  // dynamic routing, 3 iterations (fixed sequence: graph-capture safe)
  zero_kernel<<<dim3(45), 256, 0, stream>>>(bij, 11520);
  for (int it = 0; it < 3; ++it) {
    softmax_routes_kernel<<<dim3(10), 256, 0, stream>>>(bij, cij);
    route_sv_kernel<<<dim3(160), 256, 0, stream>>>(uhat, cij, vout);
    if (it < 2)
      route_bupdate_kernel<<<dim3(45), 256, 0, stream>>>(uhat, vout, bij);
  }

  classifier_kernel<<<dim3(2560), 128, 0, stream>>>(vout, cls_w1, cls_b1,
                                                    cls_w2, cls_b2, cls);
}
